// PointnetPP_73589969649777
// MI455X (gfx1250) — compile-verified, matrix operand order checked by reading, not run on hardware
//
#include <hip/hip_runtime.h>
#include <hip/hip_bf16.h>
#include <math.h>

// ---------------------------------------------------------------------------
// Types for CDNA5 WMMA (wave32, 16x16x32 bf16, fp32 accumulate)
// ---------------------------------------------------------------------------
typedef __attribute__((ext_vector_type(16))) __bf16          v16bf;
typedef __attribute__((ext_vector_type(16))) unsigned short  v16us;
typedef __attribute__((ext_vector_type(8)))  float           v8f;

#define BATCH   16
#define NPTS    8192
#define NSAMP   128
#define KNN     64
#define RADIUS2 (0.4f * 0.4f)
#define NEGINF  (-1e8f)

// ---------------------------------------------------------------------------
// 1) Farthest point sampling: one workgroup per batch, LDS-resident distances.
//    Serial chain of 127 (min-update -> argmax) steps.
// ---------------------------------------------------------------------------
__global__ __launch_bounds__(1024)
void fps_kernel(const float* __restrict__ pos, int* __restrict__ fidx,
                float* __restrict__ sp) {
    __shared__ float dist[NPTS];
    __shared__ float rv[1024];
    __shared__ int   ri[1024];
    __shared__ int   s_last;
    const int b = blockIdx.x;
    const int t = threadIdx.x;
    const float* P = pos + (size_t)b * NPTS * 3;

    for (int p = t; p < NPTS; p += 1024) dist[p] = 1e10f;
    if (t == 0) {
        s_last = 0;
        fidx[b * NSAMP] = 0;
        sp[(size_t)(b * NSAMP) * 3 + 0] = P[0];
        sp[(size_t)(b * NSAMP) * 3 + 1] = P[1];
        sp[(size_t)(b * NSAMP) * 3 + 2] = P[2];
    }
    __syncthreads();

    for (int s = 1; s < NSAMP; ++s) {
        const int last = s_last;
        const float lx = P[last * 3 + 0];
        const float ly = P[last * 3 + 1];
        const float lz = P[last * 3 + 2];
        float bestV = -1.f; int bestI = 0;
        for (int p = t; p < NPTS; p += 1024) {
            float dx = P[p * 3 + 0] - lx;
            float dy = P[p * 3 + 1] - ly;
            float dz = P[p * 3 + 2] - lz;
            float d  = dx * dx + dy * dy + dz * dz;
            float dm = fminf(dist[p], d);
            dist[p] = dm;
            if (dm > bestV) { bestV = dm; bestI = p; }
        }
        rv[t] = bestV; ri[t] = bestI;
        __syncthreads();
        for (int off = 512; off > 0; off >>= 1) {
            if (t < off) {
                if (rv[t + off] > rv[t]) { rv[t] = rv[t + off]; ri[t] = ri[t + off]; }
            }
            __syncthreads();
        }
        if (t == 0) {
            const int nl = ri[0];
            s_last = nl;
            fidx[b * NSAMP + s] = nl;
            sp[(size_t)(b * NSAMP + s) * 3 + 0] = P[nl * 3 + 0];
            sp[(size_t)(b * NSAMP + s) * 3 + 1] = P[nl * 3 + 1];
            sp[(size_t)(b * NSAMP + s) * 3 + 2] = P[nl * 3 + 2];
        }
        __syncthreads();
    }
}

// ---------------------------------------------------------------------------
// 2) kNN: one block per (b,m) row; LDS-resident sq-dist + 64 argmin passes.
// ---------------------------------------------------------------------------
__global__ __launch_bounds__(256)
void knn_kernel(const float* __restrict__ pos, const float* __restrict__ sp,
                int* __restrict__ tidx, float* __restrict__ topk) {
    __shared__ float sq[NPTS];
    __shared__ float rv[256];
    __shared__ int   ri[256];
    const int b = blockIdx.x / NSAMP;
    const int m = blockIdx.x % NSAMP;
    const int t = threadIdx.x;
    const float* P = pos + (size_t)b * NPTS * 3;
    const float sx = sp[(size_t)(b * NSAMP + m) * 3 + 0];
    const float sy = sp[(size_t)(b * NSAMP + m) * 3 + 1];
    const float sz = sp[(size_t)(b * NSAMP + m) * 3 + 2];

    for (int p = t; p < NPTS; p += 256) {
        float dx = P[p * 3 + 0] - sx;
        float dy = P[p * 3 + 1] - sy;
        float dz = P[p * 3 + 2] - sz;
        sq[p] = dx * dx + dy * dy + dz * dz;
    }
    __syncthreads();

    for (int s = 0; s < KNN; ++s) {
        float bv = 1e30f; int bi = 0;
        for (int p = t; p < NPTS; p += 256) {
            float v = sq[p];
            if (v < bv) { bv = v; bi = p; }
        }
        rv[t] = bv; ri[t] = bi;
        __syncthreads();
        for (int off = 128; off > 0; off >>= 1) {
            if (t < off) {
                if (rv[t + off] < rv[t]) { rv[t] = rv[t + off]; ri[t] = ri[t + off]; }
            }
            __syncthreads();
        }
        if (t == 0) {
            tidx[(size_t)(b * NSAMP + m) * KNN + s] = ri[0];
            topk[(size_t)(b * NSAMP + m) * KNN + s] = rv[0];
            sq[ri[0]] = 1e30f;
        }
        __syncthreads();
    }
}

// ---------------------------------------------------------------------------
// 3) Gather + build stage-0 GEMM input A0 (M=131072, K padded 6 -> 32).
// ---------------------------------------------------------------------------
__global__ __launch_bounds__(256)
void build_a0(const float* __restrict__ x, const float* __restrict__ pos,
              const float* __restrict__ sp, const int* __restrict__ tidx,
              float* __restrict__ A0) {
    const int r = blockIdx.x * blockDim.x + threadIdx.x;
    if (r >= BATCH * NSAMP * KNN) return;
    const int m = (r / KNN) % NSAMP;
    const int b = r / (KNN * NSAMP);
    const int p = tidx[r];
    const float* P = pos + ((size_t)b * NPTS + p) * 3;
    const float* S = sp + ((size_t)b * NSAMP + m) * 3;
    const float* X = x + ((size_t)b * NPTS + p) * 3;
    float* o = A0 + (size_t)r * 32;
    o[0] = P[0] - S[0]; o[1] = P[1] - S[1]; o[2] = P[2] - S[2];
    o[3] = X[0];        o[4] = X[1];        o[5] = X[2];
#pragma unroll
    for (int c = 6; c < 32; ++c) o[c] = 0.f;
}

// ---------------------------------------------------------------------------
// 4) Pack fp32 weights (Kin x N, zero-padded to Kp) into bf16 wave-fragment
//    layout: contiguous 16-element (32B) chunk per lane per (nt,ki) tile.
// ---------------------------------------------------------------------------
__global__ __launch_bounds__(256)
void pack_w(const float* __restrict__ W, unsigned short* __restrict__ out,
            int Kin, int Kp, int N) {
    const int t = blockIdx.x * blockDim.x + threadIdx.x;
    if (t >= Kp * N) return;
    const int KI   = Kp >> 5;
    const int j    = t & 15;
    const int lane = (t >> 4) & 31;
    const int rest = t >> 9;
    const int ki   = rest % KI;
    const int nt   = rest / KI;
    const int k = (ki << 5) + ((lane >> 4) << 4) + j;
    const int n = (nt << 4) + (lane & 15);
    const float v = (k < Kin) ? W[(size_t)k * N + n] : 0.f;
    out[t] = __builtin_bit_cast(unsigned short, (__bf16)v);   // hw bf16 convert
}

// ---------------------------------------------------------------------------
// 5) WMMA GEMM (bf16 in, fp32 acc), templated on (K, N, BN) so the ki loop
//    fully unrolls around v_wmma and the A fragments are converted ONCE per
//    wave then reused across all N/16 tiles. Per-channel batch statistics
//    (sum, sumsq) via LDS ds_add_f32 + global float atomics.
// ---------------------------------------------------------------------------
template<int K, int N, bool BN>
__global__ __launch_bounds__(256)
void wmma_gemm_bn(const float* __restrict__ A,
                  const float* __restrict__ scl, const float* __restrict__ shf,
                  const unsigned short* __restrict__ Wp,
                  const float* __restrict__ bias,
                  float* __restrict__ Y,
                  float* __restrict__ gsum, float* __restrict__ gsq) {
    constexpr int KI = K / 32;
    constexpr int NT = N / 16;
    __shared__ float s_sum[N], s_sq[N];
    __shared__ float s_scl[K], s_shf[K];
    for (int i = threadIdx.x; i < N; i += 256) { s_sum[i] = 0.f; s_sq[i] = 0.f; }
    if (BN)
        for (int i = threadIdx.x; i < K; i += 256) { s_scl[i] = scl[i]; s_shf[i] = shf[i]; }
    __syncthreads();

    const int wave  = threadIdx.x >> 5;
    const int lane  = threadIdx.x & 31;
    const int mTile = blockIdx.x * 8 + wave;
    const int mRow  = (mTile << 4) + (lane & 15);
    const int kh    = lane >> 4;                  // K-half held by this lane

    // ---- load + fused BN/ReLU + bf16-convert all A fragments once ----
    v16bf afrag[KI];
    const float* ap0 = A + (size_t)mRow * K + (kh << 4);
#pragma unroll
    for (int ki = 0; ki < KI; ++ki) {
        const float* ap = ap0 + (ki << 5);
        v16bf a;
#pragma unroll
        for (int j = 0; j < 16; ++j) {
            float v = ap[j];
            if (BN) {
                const int kc = (ki << 5) + (kh << 4) + j;
                v = v * s_scl[kc] + s_shf[kc];
                v = v > 0.f ? v : 0.f;            // fused BN + ReLU
            }
            a[j] = (__bf16)v;                     // v_cvt_*_bf16_f32
        }
        afrag[ki] = a;
    }

    // ---- sweep N tiles, reusing the A fragments ----
    for (int nt = 0; nt < NT; ++nt) {
        if (nt + 1 < NT)                          // global_prefetch_b8 (next B tile)
            __builtin_prefetch((const void*)(Wp + ((((size_t)(nt + 1) * KI) * 32 + lane) << 4)), 0, 3);
        v8f c = {0.f, 0.f, 0.f, 0.f, 0.f, 0.f, 0.f, 0.f};
#pragma unroll
        for (int ki = 0; ki < KI; ++ki) {
            v16us bu = *(const v16us*)(Wp + ((((size_t)nt * KI + ki) * 32 + lane) << 4));
            v16bf bf = __builtin_bit_cast(v16bf, bu);
            c = __builtin_amdgcn_wmma_f32_16x16x32_bf16(
                    false, afrag[ki], false, bf, (short)0, c, false, false);
        }
        const int n = (nt << 4) + (lane & 15);
        const float bz = bias[n];
        float ls = 0.f, lq = 0.f;
#pragma unroll
        for (int j = 0; j < 8; ++j) {
            float v = c[j] + bz;
            const int mr = (mTile << 4) + (kh << 3) + j;  // D layout: lanes>=16 hold M=8..15
            Y[(size_t)mr * N + n] = v;
            ls += v; lq += v * v;
        }
        atomicAdd(&s_sum[n], ls);      // ds_add_f32
        atomicAdd(&s_sq[n],  lq);
    }
    __syncthreads();
    for (int i = threadIdx.x; i < N; i += 256) {
        atomicAdd(&gsum[i], s_sum[i]); // global_atomic_add_f32
        atomicAdd(&gsq[i],  s_sq[i]);
    }
}

// ---------------------------------------------------------------------------
// 6) BN finalize: scale = g*rsqrt(var+eps), shift = be - mean*scale.
// ---------------------------------------------------------------------------
__global__ __launch_bounds__(256)
void finalize_bn(const float* __restrict__ sum, const float* __restrict__ sq,
                 float cnt, const float* __restrict__ g,
                 const float* __restrict__ be,
                 float* __restrict__ scl, float* __restrict__ shf, int C) {
    const int c = blockIdx.x * blockDim.x + threadIdx.x;
    if (c >= C) return;
    const float mean = sum[c] / cnt;
    const float var  = sq[c] / cnt - mean * mean;
    const float s = g[c] * rsqrtf(var + 1e-5f);
    scl[c] = s;
    shf[c] = be[c] - mean * s;
}

// ---------------------------------------------------------------------------
// 7) Radius-masked max pool over k + build stage-1 input A1 (K 131 -> 160).
// ---------------------------------------------------------------------------
__global__ __launch_bounds__(256)
void pool_a1(const float* __restrict__ Y2, const float* __restrict__ scl,
             const float* __restrict__ shf, const float* __restrict__ sp,
             const float* __restrict__ topk, float* __restrict__ A1) {
    const int t = blockIdx.x * blockDim.x + threadIdx.x;
    if (t >= BATCH * NSAMP * 160) return;
    const int c = t % 160;
    const int m = (t / 160) % NSAMP;
    const int b = t / (160 * NSAMP);
    float v;
    if (c < 3) {
        v = sp[(size_t)(b * NSAMP + m) * 3 + c];
    } else if (c < 131) {
        const int cc = c - 3;
        const float s = scl[cc], h = shf[cc];
        const float* yr = Y2 + ((size_t)(b * NSAMP + m) * KNN) * 128 + cc;
        const float* tk = topk + (size_t)(b * NSAMP + m) * KNN;
        float best = NEGINF;
        for (int k = 0; k < KNN; ++k) {
            if (tk[k] <= RADIUS2) {
                float y = yr[(size_t)k * 128] * s + h;
                y = y > 0.f ? y : 0.f;
                if (y > best) best = y;
            }
        }
        v = best;
    } else {
        v = 0.f;
    }
    A1[t] = v;
}

// ---------------------------------------------------------------------------
// 8) Final BN+ReLU + max over the 128 rows per batch -> (B,1,512).
// ---------------------------------------------------------------------------
__global__ __launch_bounds__(256)
void final_max(const float* __restrict__ Z2, const float* __restrict__ scl,
               const float* __restrict__ shf, float* __restrict__ out) {
    const int t = blockIdx.x * blockDim.x + threadIdx.x;
    if (t >= BATCH * 512) return;
    const int c = t % 512;
    const int b = t / 512;
    const float s = scl[c], h = shf[c];
    const float* z = Z2 + ((size_t)b * NSAMP) * 512 + c;
    float best = -3.4e38f;
    for (int r = 0; r < NSAMP; ++r) {
        float y = z[(size_t)r * 512] * s + h;
        y = y > 0.f ? y : 0.f;
        if (y > best) best = y;
    }
    out[t] = best;
}

// ---------------------------------------------------------------------------
// 9) Zero per-call accumulators (BN stats) + the pos_out tail of d_out.
// ---------------------------------------------------------------------------
__global__ __launch_bounds__(256)
void zero_kernel(float* __restrict__ stats, int nStats,
                 float* __restrict__ tail, int nTail) {
    const int t = blockIdx.x * blockDim.x + threadIdx.x;
    if (t < nStats) stats[t] = 0.f;
    if (t < nTail)  tail[t]  = 0.f;
}

// ---------------------------------------------------------------------------
// Host-side orchestration (all launches on `stream`, graph-capture safe).
// ---------------------------------------------------------------------------
extern "C" void kernel_launch(void* const* d_in, const int* in_sizes, int n_in,
                              void* d_out, int out_size, void* d_ws, size_t ws_size,
                              hipStream_t stream) {
    (void)in_sizes; (void)n_in; (void)out_size; (void)ws_size;
    const float* x   = (const float*)d_in[0];
    const float* pos = (const float*)d_in[1];
    const float *W[6], *bb[6], *gg[6], *be[6];
    for (int l = 0; l < 6; ++l) {
        const int base = 2 + l * 4;
        W[l]  = (const float*)d_in[base + 0];
        bb[l] = (const float*)d_in[base + 1];
        gg[l] = (const float*)d_in[base + 2];
        be[l] = (const float*)d_in[base + 3];
    }

    // Layer configs: {Kin, Kpadded, N}
    const int Kin[6] = {6, 64, 64, 131, 128, 128};
    const int Kp[6]  = {32, 64, 64, 160, 128, 128};
    const int Nn[6]  = {64, 64, 128, 128, 128, 512};

    // Workspace carve-out (float units, 256B aligned chunks).
    float* ws = (float*)d_ws;
    size_t off = 0;
    auto alloc = [&](size_t nfloats) {
        float* p = ws + off;
        off += (nfloats + 63) & ~(size_t)63;
        return p;
    };

    // BN stat accumulators first (contiguous -> one zero pass).
    float* statsBase = ws + off;
    float *sum[6], *sq[6];
    for (int l = 0; l < 6; ++l) { sum[l] = alloc(Nn[l]); sq[l] = alloc(Nn[l]); }
    const int nStats = (int)(ws + off - statsBase);
    float *scl[6], *shf[6];
    for (int l = 0; l < 6; ++l) { scl[l] = alloc(Nn[l]); shf[l] = alloc(Nn[l]); }

    int*   fidx = (int*)alloc(BATCH * NSAMP);
    float* sp   = alloc(BATCH * NSAMP * 3);
    int*   tidx = (int*)alloc(BATCH * NSAMP * KNN);
    float* topk = alloc(BATCH * NSAMP * KNN);
    unsigned short* Wp[6];
    for (int l = 0; l < 6; ++l)
        Wp[l] = (unsigned short*)alloc(((size_t)Kp[l] * Nn[l] + 1) / 2);

    float* A0 = alloc((size_t)BATCH * NSAMP * KNN * 32);
    float* Y0 = alloc((size_t)BATCH * NSAMP * KNN * 64);
    float* Y1 = alloc((size_t)BATCH * NSAMP * KNN * 64);
    float* Y2 = alloc((size_t)BATCH * NSAMP * KNN * 128);
    float* A1 = alloc((size_t)BATCH * NSAMP * 160);
    float* Z0 = alloc((size_t)BATCH * NSAMP * 128);
    float* Z1 = alloc((size_t)BATCH * NSAMP * 128);
    float* Z2 = alloc((size_t)BATCH * NSAMP * 512);

    float* out = (float*)d_out;                  // [0,8192) global_x, [8192,8240) pos_out

    // --- zero stats + pos_out tail ---
    zero_kernel<<<(nStats + 255) / 256, 256, 0, stream>>>(statsBase, nStats,
                                                          out + BATCH * 512, BATCH * 3);
    // --- FPS + kNN + gather ---
    fps_kernel<<<BATCH, 1024, 0, stream>>>(pos, fidx, sp);
    knn_kernel<<<BATCH * NSAMP, 256, 0, stream>>>(pos, sp, tidx, topk);
    build_a0<<<(BATCH * NSAMP * KNN + 255) / 256, 256, 0, stream>>>(x, pos, sp, tidx, A0);

    // --- pack all weights into bf16 fragments ---
    for (int l = 0; l < 6; ++l) {
        const int tot = Kp[l] * Nn[l];
        pack_w<<<(tot + 255) / 256, 256, 0, stream>>>(W[l], Wp[l], Kin[l], Kp[l], Nn[l]);
    }

    const float cnt0 = (float)(BATCH * NSAMP * KNN);   // 131072
    const float cnt1 = (float)(BATCH * NSAMP);         // 2048
    const int grid0 = (BATCH * NSAMP * KNN) / 128;     // 1024 blocks (8 waves x 16 rows)
    const int grid1 = (BATCH * NSAMP) / 128;           // 16 blocks

    // --- stage 0: three WMMA conv layers with interleaved BN finalize ---
    wmma_gemm_bn<32, 64, false><<<grid0, 256, 0, stream>>>(
        A0, scl[0], shf[0], Wp[0], bb[0], Y0, sum[0], sq[0]);
    finalize_bn<<<1, 256, 0, stream>>>(sum[0], sq[0], cnt0, gg[0], be[0], scl[0], shf[0], 64);

    wmma_gemm_bn<64, 64, true><<<grid0, 256, 0, stream>>>(
        Y0, scl[0], shf[0], Wp[1], bb[1], Y1, sum[1], sq[1]);
    finalize_bn<<<1, 256, 0, stream>>>(sum[1], sq[1], cnt0, gg[1], be[1], scl[1], shf[1], 64);

    wmma_gemm_bn<64, 128, true><<<grid0, 256, 0, stream>>>(
        Y1, scl[1], shf[1], Wp[2], bb[2], Y2, sum[2], sq[2]);
    finalize_bn<<<1, 256, 0, stream>>>(sum[2], sq[2], cnt0, gg[2], be[2], scl[2], shf[2], 128);

    // --- radius-masked max pool + build stage-1 input ---
    pool_a1<<<(BATCH * NSAMP * 160 + 255) / 256, 256, 0, stream>>>(
        Y2, scl[2], shf[2], sp, topk, A1);

    // --- stage 1: three WMMA conv layers ---
    wmma_gemm_bn<160, 128, false><<<grid1, 256, 0, stream>>>(
        A1, scl[3], shf[3], Wp[3], bb[3], Z0, sum[3], sq[3]);
    finalize_bn<<<1, 256, 0, stream>>>(sum[3], sq[3], cnt1, gg[3], be[3], scl[3], shf[3], 128);

    wmma_gemm_bn<128, 128, true><<<grid1, 256, 0, stream>>>(
        Z0, scl[3], shf[3], Wp[4], bb[4], Z1, sum[4], sq[4]);
    finalize_bn<<<1, 256, 0, stream>>>(sum[4], sq[4], cnt1, gg[4], be[4], scl[4], shf[4], 128);

    wmma_gemm_bn<128, 512, true><<<grid1, 256, 0, stream>>>(
        Z1, scl[4], shf[4], Wp[5], bb[5], Z2, sum[5], sq[5]);
    finalize_bn<<<2, 256, 0, stream>>>(sum[5], sq[5], cnt1, gg[5], be[5], scl[5], shf[5], 512);

    // --- final BN+ReLU + max over rows -> (B,1,512) ---
    final_max<<<(BATCH * 512 + 255) / 256, 256, 0, stream>>>(Z2, scl[5], shf[5], out);
}
